// FlexAttentionCPB_18554258718869
// MI455X (gfx1250) — matrix operand order repaired and checked by reading, not compiled, not run on hardware
//
#include <hip/hip_runtime.h>
#include <hip/hip_bf16.h>

typedef __attribute__((ext_vector_type(16))) _Float16 v16h;
typedef __attribute__((ext_vector_type(8)))  float    v8f;

#define B_N   8
#define H_N   16
#define S_LEN 1028
#define DH    64
#define L2_N  3969
#define HID_N 32
#define S2PAD 1056   // padded key-stride of transposed V (2112 B rows, 16B aligned)

// LDS strides (in halves), padded against bank conflicts
#define KSTR 72   // K tile rows: 32 keys x 64 dims (+8 pad)
#define VSTR 40   // V^T rows:    64 dims x 32 keys (+8 pad)
#define PSTR 36   // P tile rows: 16 x 32 keys (+4 pad)
#define QSTR 72   // Q stage:     64 rows x 64 dims (+8 pad)

static __device__ __forceinline__ unsigned lds_lo32(const void* p) {
    // ISA 10.2: the low 32 bits of an LDS-aperture flat address are its LDS offset
    return (unsigned)(unsigned long long)(uintptr_t)p;
}
template <int OFF>
static __device__ __forceinline__ void async_b128(unsigned lds, unsigned long long g) {
    asm volatile("global_load_async_to_lds_b128 %0, %1, off offset:%2"
                 :: "v"(lds), "v"(g), "i"(OFF) : "memory");
}
template <int N>
static __device__ __forceinline__ void wait_async_le() {
#if __has_builtin(__builtin_amdgcn_s_wait_asynccnt)
    __builtin_amdgcn_s_wait_asynccnt(N);
#else
    asm volatile("s_wait_asynccnt %0" :: "i"(N) : "memory");
#endif
}

// ---------------------------------------------------------------------------
// Kernel 1: CPB MLP  bt[l2][h] = gelu_exact(rel @ W1 + b1) @ W2
// ---------------------------------------------------------------------------
__global__ void cpb_mlp_kernel(const float* __restrict__ rel,
                               const float* __restrict__ W1,
                               const float* __restrict__ b1,
                               const float* __restrict__ W2,
                               float* __restrict__ bt) {
    int i = blockIdx.x * blockDim.x + threadIdx.x;
    if (i >= L2_N) return;
    float x0 = rel[i * 2 + 0];
    float x1 = rel[i * 2 + 1];
    float acc[H_N];
#pragma unroll
    for (int h = 0; h < H_N; ++h) acc[h] = 0.0f;
    for (int j = 0; j < HID_N; ++j) {
        float pre = x0 * W1[j] + x1 * W1[HID_N + j] + b1[j];
        float g = 0.5f * pre * (1.0f + erff(pre * 0.70710678118654752f)); // exact GELU
#pragma unroll
        for (int h = 0; h < H_N; ++h) acc[h] += g * W2[j * H_N + h];
    }
#pragma unroll
    for (int h = 0; h < H_N; ++h) bt[i * H_N + h] = acc[h];
}

// ---------------------------------------------------------------------------
// Kernel 1b: one-time f32->f16 convert of K, and convert+transpose of V
//   kh [bh][s][d]      (rows of 128B, contiguous)
//   vth[bh][d][s_pad]  (rows of 2112B, 16B-aligned chunks for async copy)
// ---------------------------------------------------------------------------
__global__ __launch_bounds__(256, 1)
void convert_kv_kernel(const float* __restrict__ k, const float* __restrict__ v,
                       _Float16* __restrict__ kh, _Float16* __restrict__ vth) {
    __shared__ _Float16 T[64 * 72];
    const int bh = blockIdx.y;
    const int s0 = blockIdx.x * 64;
    const int tid = threadIdx.x;
    const float*  kp  = k   + (size_t)bh * S_LEN * DH;
    const float*  vp  = v   + (size_t)bh * S_LEN * DH;
    _Float16*     khp = kh  + (size_t)bh * S_LEN * DH;
    _Float16*     vtp = vth + (size_t)bh * DH * S2PAD;
    for (int e = 0; e < 16; ++e) {
        int idx = e * 256 + tid;
        int r = idx >> 6, d = idx & 63;
        int gs = s0 + r;
        if (gs < S_LEN) {
            size_t off = (size_t)gs * DH + d;
            khp[off]      = (_Float16)kp[off];
            T[r * 72 + d] = (_Float16)vp[off];
        }
    }
    __syncthreads();
    for (int e = 0; e < 16; ++e) {
        int idx = e * 256 + tid;
        int d = idx >> 6, j = idx & 63;
        int gs = s0 + j;
        if (gs < S_LEN) vtp[(size_t)d * S2PAD + gs] = T[j * 72 + d];
    }
}

// ---------------------------------------------------------------------------
// Kernel 2: flash attention with CPB, f16 WMMA / f32 accumulate.
// ASYNC=true : double-buffered global_load_async_to_lds_b128 prefetch pipeline
//              (tile i+1 in flight while tile i computes; s_wait_asynccnt 4).
// ASYNC=false: direct f32 load + convert (fallback when ws too small).
// grid = (ceil(S/64), H, B), block = 128 (4 waves, 16 q-rows per wave)
// ---------------------------------------------------------------------------
template <bool ASYNC>
__global__ __launch_bounds__(128, 1)
void flexattn_cpb_kernel(const float* __restrict__ q,
                         const float* __restrict__ k,
                         const float* __restrict__ v,
                         const _Float16* __restrict__ kh,
                         const _Float16* __restrict__ vth,
                         const int*   __restrict__ idx_table,
                         const float* __restrict__ bt,
                         const int*   __restrict__ Rp,
                         float* __restrict__ out) {
    __shared__ _Float16 Kl[2][32 * KSTR];     // double-buffered [key][dim]
    __shared__ _Float16 Vt[2][64 * VSTR];     // double-buffered [dim][key]
    __shared__ _Float16 Pl[4][16 * PSTR];     // per-wave P tile [row][key]
    __shared__ _Float16 Qs[64 * QSTR];        // Q stage [row][dim]

    const int tid  = threadIdx.x;
    const int wave = tid >> 5;
    const int lane = tid & 31;
    const int m    = lane & 15;
    const int hi   = lane >> 4;
    const int h    = blockIdx.y;
    const int b    = blockIdx.z;
    const int qbase = blockIdx.x * 64;
    const int R    = Rp[0];
    const float scale = 0.125f;               // Dh^-0.5

    const int    bh_lin = b * H_N + h;
    const size_t bh     = (size_t)bh_lin * S_LEN;
    const float* qp = q + bh * DH;
    const float* kp = k + bh * DH;
    const float* vp = v + bh * DH;

    // ---- per-thread async-copy geometry (loop-invariant); tile base advances
    //      by constants: K += 32*64*2 = 4096 B, V^T += 32*2 = 64 B ----
    unsigned long long kaddr = 0, vaddr = 0;
    unsigned kl0 = 0, kl1 = 0, vl0 = 0, vl1 = 0;
    if constexpr (ASYNC) {
        const _Float16* khb = kh  + bh * DH;
        const _Float16* vtb = vth + (size_t)bh_lin * DH * S2PAD;
        int key = tid >> 3, o  = tid & 7;   // K chunk: 8 x 16B per 128B row
        int d   = tid >> 2, o2 = tid & 3;   // V chunk: 4 x 16B per 64B segment
        kaddr = (unsigned long long)(uintptr_t)(khb + key * DH + o * 8);
        vaddr = (unsigned long long)(uintptr_t)(vtb + (size_t)d * S2PAD + o2 * 8);
        kl0 = lds_lo32(&Kl[0][0]) + (unsigned)(key * (KSTR * 2) + o * 16);
        kl1 = lds_lo32(&Kl[1][0]) + (unsigned)(key * (KSTR * 2) + o * 16);
        vl0 = lds_lo32(&Vt[0][0]) + (unsigned)(d * (VSTR * 2) + o2 * 16);
        vl1 = lds_lo32(&Vt[1][0]) + (unsigned)(d * (VSTR * 2) + o2 * 16);
    }
    auto issue = [&](int parity) {            // 4 async ops per wave per tile
        unsigned kl = parity ? kl1 : kl0;
        unsigned vl = parity ? vl1 : vl0;
        async_b128<0>(kl, kaddr);                                 // keys 0-15 half
        async_b128<2048>(kl + 16 * (KSTR * 2), kaddr);            // keys 16-31 half
        async_b128<0>(vl, vaddr);                                 // dims 0-31 half
        async_b128<32 * S2PAD * 2>(vl + 32 * (VSTR * 2), vaddr);  // dims 32-63 half
        kaddr += 32 * DH * 2;
        vaddr += 32 * 2;
    };
    if constexpr (ASYNC) issue(0);            // prefetch tile 0 behind Q staging

    // ---- stage Q (pre-scaled) into LDS as f16 ----
    for (int e = 0; e < 32; ++e) {
        int idx = e * 128 + tid;
        int row = idx >> 6, d = idx & 63;
        int gq = qbase + row; if (gq >= S_LEN) gq = S_LEN - 1;
        Qs[row * QSTR + d] = (_Float16)(qp[(size_t)gq * DH + d] * scale);
    }
    __syncthreads();

    // ---- Q A-layout regs (lane m: elems 0-15 -> K=hi*16+e) ----
    v16h aq0, aq1;
    {
        const int r0 = wave * 16 + m;
#pragma unroll
        for (int e = 0; e < 8; ++e) {
            aq0[e]     = Qs[r0 * QSTR +      hi * 16 + e];
            aq0[8 + e] = Qs[r0 * QSTR +  8 + hi * 16 + e];
            aq1[e]     = Qs[r0 * QSTR + 32 + hi * 16 + e];
            aq1[8 + e] = Qs[r0 * QSTR + 40 + hi * 16 + e];
        }
    }

    v8f o0 = {}, o1 = {}, o2 = {}, o3 = {};
    float mi[8], li[8];
#pragma unroll
    for (int r = 0; r < 8; ++r) { mi[r] = -3.0e38f; li[r] = 0.0f; }

    const int nkt = (S_LEN + 31) / 32;        // 33 key tiles
    for (int kb = 0; kb < nkt; ++kb) {
        const int kbase = kb * 32;
        __syncthreads();   // all waves done with buf[(kb+1)&1] from iter kb-1
        if constexpr (ASYNC) {
            if (kb + 1 < nkt) { issue((kb + 1) & 1); wait_async_le<4>(); }
            else              { wait_async_le<0>(); }
        } else {
            for (int e = 0; e < 16; ++e) {
                int idx = e * 128 + tid;
                int key = idx >> 6, d = idx & 63;
                int gk = kbase + key; if (gk >= S_LEN) gk = S_LEN - 1;
                size_t off = (size_t)gk * DH + d;
                Kl[kb & 1][key * KSTR + d] = (_Float16)kp[off];
                Vt[kb & 1][d * VSTR + key] = (_Float16)vp[off];
            }
        }
        __syncthreads();   // tile kb visible to all waves
        const _Float16* __restrict__ KT = &Kl[kb & 1][0];
        const _Float16* __restrict__ VT = &Vt[kb & 1][0];

        // ---- scores: two 16-key sub-tiles, Dh=64 = 2 WMMAs each ----
        v8f s0 = {}, s1 = {};
        {
            v16h bk;
#pragma unroll
            for (int e = 0; e < 16; ++e) bk[e] = KT[m * KSTR + hi * 16 + e];
            s0 = __builtin_amdgcn_wmma_f32_16x16x32_f16(false, aq0, false, bk, (short)0, s0, false, false);
#pragma unroll
            for (int e = 0; e < 16; ++e) bk[e] = KT[m * KSTR + 32 + hi * 16 + e];
            s0 = __builtin_amdgcn_wmma_f32_16x16x32_f16(false, aq1, false, bk, (short)0, s0, false, false);
#pragma unroll
            for (int e = 0; e < 16; ++e) bk[e] = KT[(16 + m) * KSTR + hi * 16 + e];
            s1 = __builtin_amdgcn_wmma_f32_16x16x32_f16(false, aq0, false, bk, (short)0, s1, false, false);
#pragma unroll
            for (int e = 0; e < 16; ++e) bk[e] = KT[(16 + m) * KSTR + 32 + hi * 16 + e];
            s1 = __builtin_amdgcn_wmma_f32_16x16x32_f16(false, aq1, false, bk, (short)0, s1, false, false);
        }

        // ---- bias gather (32-bit offsets) + online softmax ----
        const int key0 = kbase + m, key1 = kbase + 16 + m;
        const bool k0ok = key0 < S_LEN, k1ok = key1 < S_LEN;
        float alpha[8];
#pragma unroll
        for (int r = 0; r < 8; ++r) {
            float x0 = s0[r], x1 = s1[r];
            const int qg = qbase + wave * 16 + r + 8 * hi;
            if (qg < S_LEN && qg >= R) {
                const int rowoff = qg * S_LEN;
                if (k0ok && key0 >= R) x0 += bt[idx_table[rowoff + key0] * H_N + h];
                if (k1ok && key1 >= R) x1 += bt[idx_table[rowoff + key1] * H_N + h];
            }
            if (!k0ok) x0 = -3.0e38f;
            if (!k1ok) x1 = -3.0e38f;
            float rm = fmaxf(x0, x1);
            rm = fmaxf(rm, __shfl_xor(rm, 1, 32));
            rm = fmaxf(rm, __shfl_xor(rm, 2, 32));
            rm = fmaxf(rm, __shfl_xor(rm, 4, 32));
            rm = fmaxf(rm, __shfl_xor(rm, 8, 32));
            float mnew = fmaxf(mi[r], rm);
            float a  = __expf(mi[r] - mnew);
            float p0 = k0ok ? __expf(x0 - mnew) : 0.0f;
            float p1 = k1ok ? __expf(x1 - mnew) : 0.0f;
            float rs = p0 + p1;
            rs += __shfl_xor(rs, 1, 32);
            rs += __shfl_xor(rs, 2, 32);
            rs += __shfl_xor(rs, 4, 32);
            rs += __shfl_xor(rs, 8, 32);
            li[r] = li[r] * a + rs;
            mi[r] = mnew;
            alpha[r] = a;
            _Float16* pw = &Pl[wave][(r + 8 * hi) * PSTR];
            pw[m]      = (_Float16)p0;
            pw[16 + m] = (_Float16)p1;
        }
        __syncthreads();   // cross-lane P visibility (C-layout -> A-layout)

        // ---- P back in A-layout; rescale O; accumulate P @ V ----
        v16h ap;
        {
            const _Float16* pr = &Pl[wave][m * PSTR];
#pragma unroll
            for (int e = 0; e < 8; ++e) {
                ap[e]     = pr[hi * 16 + e];
                ap[8 + e] = pr[8 + hi * 16 + e];
            }
        }
#pragma unroll
        for (int r = 0; r < 8; ++r) {
            o0[r] *= alpha[r]; o1[r] *= alpha[r];
            o2[r] *= alpha[r]; o3[r] *= alpha[r];
        }
        v16h bv;
#pragma unroll
        for (int e = 0; e < 16; ++e) bv[e] = VT[(0 * 16 + m) * VSTR + hi * 16 + e];
        o0 = __builtin_amdgcn_wmma_f32_16x16x32_f16(false, ap, false, bv, (short)0, o0, false, false);
#pragma unroll
        for (int e = 0; e < 16; ++e) bv[e] = VT[(1 * 16 + m) * VSTR + hi * 16 + e];
        o1 = __builtin_amdgcn_wmma_f32_16x16x32_f16(false, ap, false, bv, (short)0, o1, false, false);
#pragma unroll
        for (int e = 0; e < 16; ++e) bv[e] = VT[(2 * 16 + m) * VSTR + hi * 16 + e];
        o2 = __builtin_amdgcn_wmma_f32_16x16x32_f16(false, ap, false, bv, (short)0, o2, false, false);
#pragma unroll
        for (int e = 0; e < 16; ++e) bv[e] = VT[(3 * 16 + m) * VSTR + hi * 16 + e];
        o3 = __builtin_amdgcn_wmma_f32_16x16x32_f16(false, ap, false, bv, (short)0, o3, false, false);
    }

    // ---- normalize and store ----
    float* op = out + bh * DH;
#pragma unroll
    for (int r = 0; r < 8; ++r) {
        int qg = qbase + wave * 16 + r + 8 * hi;
        if (qg < S_LEN) {
            float inv = 1.0f / li[r];
            size_t base = (size_t)qg * DH;
            op[base +  0 + m] = o0[r] * inv;
            op[base + 16 + m] = o1[r] * inv;
            op[base + 32 + m] = o2[r] * inv;
            op[base + 48 + m] = o3[r] * inv;
        }
    }
}

// ---------------------------------------------------------------------------
extern "C" void kernel_launch(void* const* d_in, const int* in_sizes, int n_in,
                              void* d_out, int out_size, void* d_ws, size_t ws_size,
                              hipStream_t stream) {
    const float* q    = (const float*)d_in[0];
    const float* kk   = (const float*)d_in[1];
    const float* vv   = (const float*)d_in[2];
    const float* rel  = (const float*)d_in[3];
    const float* W1   = (const float*)d_in[4];
    const float* b1   = (const float*)d_in[5];
    const float* W2   = (const float*)d_in[6];
    const int*   idxt = (const int*)d_in[7];
    const int*   Rp   = (const int*)d_in[8];
    float* out = (float*)d_out;

    (void)in_sizes; (void)n_in; (void)out_size;

    // workspace layout: bt | kh | vth  (K-tail async reads from the last head
    // spill harmlessly into the vth region, so kh must precede vth)
    const size_t btsz  = (size_t)L2_N * H_N * sizeof(float);
    const size_t khoff = (btsz + 255) & ~(size_t)255;
    const size_t khsz  = (size_t)B_N * H_N * S_LEN * DH * sizeof(_Float16);
    const size_t vtoff = (khoff + khsz + 255) & ~(size_t)255;
    const size_t vtsz  = (size_t)B_N * H_N * DH * S2PAD * sizeof(_Float16);
    const bool fast = ws_size >= vtoff + vtsz;

    float* bt = (float*)d_ws;
    cpb_mlp_kernel<<<(L2_N + 127) / 128, 128, 0, stream>>>(rel, W1, b1, W2, bt);

    dim3 grid((S_LEN + 63) / 64, H_N, B_N);
    if (fast) {
        _Float16* kh  = (_Float16*)((char*)d_ws + khoff);
        _Float16* vth = (_Float16*)((char*)d_ws + vtoff);
        dim3 cgrid((S_LEN + 63) / 64, B_N * H_N);
        convert_kv_kernel<<<cgrid, 256, 0, stream>>>(kk, vv, kh, vth);
        flexattn_cpb_kernel<true><<<grid, 128, 0, stream>>>(
            q, kk, vv, kh, vth, idxt, bt, Rp, out);
    } else {
        flexattn_cpb_kernel<false><<<grid, 128, 0, stream>>>(
            q, kk, vv, nullptr, nullptr, idxt, bt, Rp, out);
    }
}
